// MultiHeadAttention_18098992185448
// MI455X (gfx1250) — matrix operand order repaired and checked by reading, not compiled, not run on hardware
//
#include <hip/hip_runtime.h>
#include <hip/hip_bf16.h>

#define EMBED 1024
#define HEADS 16
#define DH    64
#define BB    2
#define SS    2048

typedef __attribute__((ext_vector_type(16))) __bf16 v16bf;
typedef __attribute__((ext_vector_type(8)))  float  v8f;

__device__ __forceinline__ v8f wmma_bf16(v16bf a, v16bf b, v8f c) {
  // (neg_a, A, neg_b, B, c_mod, C, reuse_a, reuse_b)
  return __builtin_amdgcn_wmma_f32_16x16x32_bf16(false, a, false, b, (short)0, c,
                                                 false, false);
}

// A operand (M=16 x K=32, 16-bit): lane -> M (lane&15); element e -> K = half*16 + e
template <typename T>
__device__ __forceinline__ v16bf load_a(const T* base, int ld) {
  const int lm = threadIdx.x & 15, hf = (threadIdx.x >> 4) & 1;
  const T* row = base + (size_t)lm * ld + hf * 16;
  v16bf a;
#pragma unroll
  for (int e = 0; e < 16; ++e) a[e] = (__bf16)(float)row[e];
  return a;
}

// B operand (K=32 x N=16): lane -> N, element e -> K = half*16 + e.
// Source is row-major [N][K] (i.e. holds B^T): 16 contiguous loads per lane.
template <typename T>
__device__ __forceinline__ v16bf load_b_nmajor(const T* base, size_t ld) {
  const int lm = threadIdx.x & 15, hf = (threadIdx.x >> 4) & 1;
  const T* row = base + (size_t)lm * ld + hf * 16;
  v16bf b;
#pragma unroll
  for (int e = 0; e < 16; ++e) b[e] = (__bf16)(float)row[e];
  return b;
}

// ---------------- Kernel 1: per-head Q/K/V projection -----------------------
// x: [B,S,EMBED] fp32 viewed as [B*S*H, 64] rows. One wave = one (b,s) tile of
// 16 rows (h=0..15) x 64 cols. Y = X @ W^T + bias, output bf16.
// VT=false: write [B,H,S,64] (row-major per head, used for Q and K).
// VT=true : write [B,H,64,S] (transposed, so P@V B-fragments are contiguous).
template <bool VT>
__global__ __launch_bounds__(256) void proj_kernel(const float* __restrict__ x,
                                                   const float* __restrict__ W,
                                                   const float* __restrict__ bias,
                                                   __bf16* __restrict__ out) {
  const int lm = threadIdx.x & 15, hf = (threadIdx.x >> 4) & 1;
  const int wave = threadIdx.x >> 5;
  const int rt = blockIdx.x * 8 + wave;  // 0..B*S-1
  const int b = rt >> 11;                // rt / SS
  const int s = rt & (SS - 1);

  const float* xt = x + (size_t)rt * (16 * DH);
  v16bf a0 = load_a(xt, DH);
  v16bf a1 = load_a(xt + 32, DH);

#pragma unroll
  for (int nt = 0; nt < 4; ++nt) {
    const int d = nt * 16 + lm;
    const float bval = bias[d];
    v8f acc;
#pragma unroll
    for (int r = 0; r < 8; ++r) acc[r] = bval;
    // B = W^T ; W is row-major [d][k] = [N][K]
    v16bf b0 = load_b_nmajor(W + (size_t)(nt * 16) * DH, DH);
    v16bf b1 = load_b_nmajor(W + (size_t)(nt * 16) * DH + 32, DH);
    acc = wmma_bf16(a0, b0, acc);
    acc = wmma_bf16(a1, b1, acc);
#pragma unroll
    for (int r = 0; r < 8; ++r) {
      const int h = hf * 8 + r;  // row M of tile == head index
      if (VT)
        out[((size_t)(b * HEADS + h) * DH + d) * SS + s] = (__bf16)acc[r];
      else
        out[((size_t)(b * HEADS + h) * SS + s) * DH + d] = (__bf16)acc[r];
    }
  }
}

// ---------------- Kernel 1b: Wo fp32 -> bf16 (done once) --------------------
__global__ __launch_bounds__(256) void cvt_w_kernel(const float* __restrict__ w,
                                                    __bf16* __restrict__ wb) {
  const size_t i = ((size_t)blockIdx.x * 256 + threadIdx.x) * 4;
  const float4 f = *(const float4*)(w + i);
  wb[i + 0] = (__bf16)f.x;
  wb[i + 1] = (__bf16)f.y;
  wb[i + 2] = (__bf16)f.z;
  wb[i + 3] = (__bf16)f.w;
}

// ---------------- Kernel 2: flash attention ---------------------------------
// One wave per (b,h, 16-row q tile). Streams kv in blocks of 32 with online
// softmax. Mask applied BEFORE the 1/sqrt(dh) scale, as in the reference.
// Vt is the transposed V projection [B,H,64,S] so its B-fragments are
// contiguous per lane.
__global__ __launch_bounds__(256) void attn_kernel(const __bf16* __restrict__ Qp,
                                                   const __bf16* __restrict__ Kp,
                                                   const __bf16* __restrict__ Vt,
                                                   const int* __restrict__ mask,
                                                   __bf16* __restrict__ ctx) {
  const int lm = threadIdx.x & 15, hf = (threadIdx.x >> 4) & 1;
  const int wave = threadIdx.x >> 5;
  __shared__ __bf16 plds[8][16 * 32];  // per-wave P staging (C/D -> A layout)
  __bf16* pw = &plds[wave][0];

  const int wid = blockIdx.x * 8 + wave;  // 0..4095
  const int qt = wid & 127;               // S/16 q tiles
  const int bh = wid >> 7;                // b*H + h
  const int b = bh >> 4;
  const int h = bh & 15;

  const __bf16* qbase = Qp + ((size_t)bh * SS + (size_t)qt * 16) * DH;
  v16bf qa0 = load_a(qbase, DH);
  v16bf qa1 = load_a(qbase + 32, DH);

  v8f o[4];
  float m[8], l[8];
#pragma unroll
  for (int nt = 0; nt < 4; ++nt)
#pragma unroll
    for (int r = 0; r < 8; ++r) o[nt][r] = 0.f;
#pragma unroll
  for (int r = 0; r < 8; ++r) { m[r] = -3.0e38f; l[r] = 0.f; }

  const int* mrow = mask + (size_t)b * SS * SS + (size_t)(qt * 16) * SS;
  const __bf16* vhead = Vt + (size_t)bh * DH * SS;  // [64][S]

  for (int kv0 = 0; kv0 < SS; kv0 += 32) {
    const __bf16* kb = Kp + ((size_t)bh * SS + kv0) * DH;
    v8f c1, c2;
#pragma unroll
    for (int r = 0; r < 8; ++r) { c1[r] = 0.f; c2[r] = 0.f; }
    // scores: B = K^T, source Kp is [kv][64] = [N][K]
    c1 = wmma_bf16(qa0, load_b_nmajor(kb, DH), c1);
    c1 = wmma_bf16(qa1, load_b_nmajor(kb + 32, DH), c1);
    c2 = wmma_bf16(qa0, load_b_nmajor(kb + 16 * DH, DH), c2);
    c2 = wmma_bf16(qa1, load_b_nmajor(kb + 16 * DH + 32, DH), c2);

#pragma unroll
    for (int r = 0; r < 8; ++r) {
      const int* mp = mrow + (size_t)(hf * 8 + r) * SS + kv0;
      float s1 = c1[r]; if (mp[lm] == 0)      s1 = -1.0e20f; s1 *= 0.125f;
      float s2 = c2[r]; if (mp[16 + lm] == 0) s2 = -1.0e20f; s2 *= 0.125f;
      // row max across the 16 lanes of this half (row = hf*8+r)
      float tm = fmaxf(s1, s2);
#pragma unroll
      for (int off = 1; off < 16; off <<= 1) tm = fmaxf(tm, __shfl_xor(tm, off, 32));
      const float mn = fmaxf(m[r], tm);
      const float alpha = __expf(m[r] - mn);
      m[r] = mn;
      const float p1 = __expf(s1 - mn), p2 = __expf(s2 - mn);
      float rs = p1 + p2;
#pragma unroll
      for (int off = 1; off < 16; off <<= 1) rs += __shfl_xor(rs, off, 32);
      l[r] = l[r] * alpha + rs;
#pragma unroll
      for (int nt = 0; nt < 4; ++nt) o[nt][r] *= alpha;
      // stage P (16x32, row-major) in LDS for A-operand reload
      pw[(hf * 8 + r) * 32 + lm]      = (__bf16)p1;
      pw[(hf * 8 + r) * 32 + 16 + lm] = (__bf16)p2;
    }
    v16bf pa = load_a(pw, 32);  // LDS ops are in-order within a wave
    // P@V : B = V[kv][d]; Vt holds V^T row-major [d][kv] = [N][K] -> contiguous
#pragma unroll
    for (int nt = 0; nt < 4; ++nt)
      o[nt] = wmma_bf16(pa, load_b_nmajor(vhead + (size_t)(nt * 16) * SS + kv0,
                                          (size_t)SS),
                        o[nt]);
  }

#pragma unroll
  for (int r = 0; r < 8; ++r) {
    const int qrow = qt * 16 + hf * 8 + r;
    const float inv = 1.0f / l[r];
    const size_t obase = ((size_t)(b * SS + qrow)) * EMBED + (size_t)h * DH;
#pragma unroll
    for (int nt = 0; nt < 4; ++nt)
      ctx[obase + nt * 16 + lm] = (__bf16)(o[nt][r] * inv);
  }
}

// ---------------- Kernel 3: output projection -------------------------------
// out[n,e] = ctx[n,:] . Wo[e,:] + bo[e] ; 4096 x 1024 x 1024 GEMM, fp32 out.
// One wave computes a 32-row x 64-col tile: two A fragments share every
// B (weight) fragment, halving weight traffic per FLOP. Wob is pre-converted
// bf16 so the K-loop is pure b128 loads + WMMA.
__global__ __launch_bounds__(256) void outproj_kernel(const __bf16* __restrict__ ctx,
                                                      const __bf16* __restrict__ Wob,
                                                      const float* __restrict__ bo,
                                                      float* __restrict__ out) {
  const int lm = threadIdx.x & 15, hf = (threadIdx.x >> 4) & 1;
  const int wave = threadIdx.x >> 5;
  const int wid = blockIdx.x * 8 + wave;  // 0..2047
  const int ct = wid & 15;                // 16 col tiles of 64
  const int rt = wid >> 4;                // 128 row tiles of 32

  v8f acc[2][4];
#pragma unroll
  for (int nt = 0; nt < 4; ++nt) {
    const float bval = bo[ct * 64 + nt * 16 + lm];
#pragma unroll
    for (int r = 0; r < 8; ++r) { acc[0][nt][r] = bval; acc[1][nt][r] = bval; }
  }
  const __bf16* abase = ctx + (size_t)rt * 32 * EMBED;
  for (int kc = 0; kc < EMBED; kc += 32) {
    v16bf a0 = load_a(abase + kc, EMBED);
    v16bf a1 = load_a(abase + (size_t)16 * EMBED + kc, EMBED);
#pragma unroll
    for (int nt = 0; nt < 4; ++nt) {
      v16bf bfrag =
          load_b_nmajor(Wob + (size_t)(ct * 64 + nt * 16) * EMBED + kc,
                        (size_t)EMBED);
      acc[0][nt] = wmma_bf16(a0, bfrag, acc[0][nt]);
      acc[1][nt] = wmma_bf16(a1, bfrag, acc[1][nt]);
    }
  }
#pragma unroll
  for (int mt = 0; mt < 2; ++mt)
#pragma unroll
    for (int r = 0; r < 8; ++r) {
      const int row = rt * 32 + mt * 16 + hf * 8 + r;
#pragma unroll
      for (int nt = 0; nt < 4; ++nt)
        out[(size_t)row * EMBED + ct * 64 + nt * 16 + lm] = acc[mt][nt][r];
    }
}

extern "C" void kernel_launch(void* const* d_in, const int* in_sizes, int n_in,
                              void* d_out, int out_size, void* d_ws, size_t ws_size,
                              hipStream_t stream) {
  const float* k  = (const float*)d_in[0];
  const float* q  = (const float*)d_in[1];
  const float* v  = (const float*)d_in[2];
  const int*  msk = (const int*)d_in[3];
  const float* Wk = (const float*)d_in[4];
  const float* bk = (const float*)d_in[5];
  const float* Wq = (const float*)d_in[6];
  const float* bq = (const float*)d_in[7];
  const float* Wv = (const float*)d_in[8];
  const float* bv = (const float*)d_in[9];
  const float* Wo = (const float*)d_in[10];
  const float* bo = (const float*)d_in[11];
  float* out = (float*)d_out;

  const size_t per = (size_t)BB * HEADS * SS * DH;  // 4M elements
  __bf16* Qp  = (__bf16*)d_ws;     // [B,H,S,64] bf16   (8 MB)
  __bf16* Kp  = Qp + per;          // [B,H,S,64]        (8 MB)
  __bf16* Vt  = Kp + per;          // [B,H,64,S]        (8 MB)
  __bf16* Ctx = Vt + per;          // [B,S,1024]        (8 MB)
  __bf16* Wob = Ctx + per;         // [1024,1024] bf16  (2 MB)

  dim3 blk(256), g(512);  // 8 waves/block
  proj_kernel<false><<<g, blk, 0, stream>>>(q, Wq, bq, Qp);
  proj_kernel<false><<<g, blk, 0, stream>>>(k, Wk, bk, Kp);
  proj_kernel<true><<<g, blk, 0, stream>>>(v, Wv, bv, Vt);
  cvt_w_kernel<<<dim3(EMBED * EMBED / (256 * 4)), blk, 0, stream>>>(Wo, Wob);
  attn_kernel<<<g, blk, 0, stream>>>(Qp, Kp, Vt, msk, Ctx);
  outproj_kernel<<<dim3(256), blk, 0, stream>>>(Ctx, Wob, bo, out);
}
